// CVDGCN_88862873355008
// MI455X (gfx1250) — matrix operand order repaired and checked by reading, not compiled, not run on hardware
//
#include <hip/hip_runtime.h>
#include <hip/hip_bf16.h>

typedef __attribute__((ext_vector_type(16))) _Float16 v16h;
typedef __attribute__((ext_vector_type(8)))  float    v8f;

#define HIDDEN 64

// ---------------------------------------------------------------------------
// deg[i] = 1.0 (self loop)
__global__ void gcn_fill1(float* __restrict__ deg, int n) {
    int i = blockIdx.x * blockDim.x + threadIdx.x;
    if (i < n) deg[i] = 1.0f;
}

// deg[dst[e]] += 1
__global__ void gcn_deg_edges(const int* __restrict__ dst, float* __restrict__ deg, int nE) {
    int e = blockIdx.x * blockDim.x + threadIdx.x;
    if (e < nE) atomicAdd(&deg[dst[e]], 1.0f);
}

// deg -> rsqrt(deg)  (deg >= 1 always, self-loops guarantee it)
__global__ void gcn_rsqrt(float* __restrict__ deg, int n) {
    int i = blockIdx.x * blockDim.x + threadIdx.x;
    if (i < n) deg[i] = rsqrtf(deg[i]);
}

// ---------------------------------------------------------------------------
// Y[n_rows x 64] = X[n_rows x K] @ W[K x 64]   (f16 WMMA, f32 accumulate)
// One wave = 16-row strip, all 64 cols. 8 waves / 256-thread block.
//
// W is pre-swizzled in LDS into B-fragment-major order so the inner loop
// fetches each v16h B operand as one contiguous 32B vector (2x ds_load_b128).
__global__ __launch_bounds__(256)
void gcn_gemm_wmma(const float* __restrict__ X, const float* __restrict__ W,
                   float* __restrict__ Y, int n_rows, int K) {
    // lB[frag][lane][j]: frag = kc*4+nt (<=16), lane<32, j<16 halves. 16 KB max.
    __shared__ _Float16 lB[16 * 32 * 16];

    const int tid   = threadIdx.x;
    const int nKc   = K >> 5;          // K-chunks of 32
    const int nFrag = nKc * 4;         // (kc, nt) fragments

    // Stage + swizzle W:  b[j] = W[kc*32 + (lane>>4)*8 + (j&7) + (j>>3)*16][nt*16 + (lane&15)]
    for (int i = tid; i < nFrag * 32 * 16; i += 256) {
        const int j    = i & 15;
        const int lane = (i >> 4) & 31;
        const int frag = i >> 9;
        const int nt   = frag & 3;
        const int kc   = frag >> 2;
        const int Kk   = kc * 32 + (lane >> 4) * 8 + (j & 7) + (j >> 3) * 16;
        const int n    = nt * 16 + (lane & 15);
        lB[i] = (_Float16)W[Kk * HIDDEN + n];
    }
    __syncthreads();

    const int wave  = tid >> 5;
    const int lane  = tid & 31;
    const int halfw = lane >> 4;       // which 16-lane half
    const int lm    = lane & 15;
    const int M0    = (blockIdx.x * 8 + wave) * 16;
    if (M0 >= n_rows) return;          // wave-uniform exit (after barrier)

    // Row used for A loads, clamped so EXEC stays all-1s for WMMA.
    int Mr = M0 + lm;
    if (Mr > n_rows - 1) Mr = n_rows - 1;
    const float* xrow = X + (size_t)Mr * K;

    v8f acc[4] = {};                   // 4 N-tiles of 16 cols

    for (int kc = 0; kc < nKc; ++kc) {
        // A fragment: 16-bit A 16x32 layout — lane holds M=lm,
        // K = kc*32 + halfw*8 + {0..7} and + 16 + {0..7} (two contiguous runs)
        const int kb = kc * 32 + halfw * 8;
        v16h a;
#pragma unroll
        for (int j = 0; j < 8; ++j) a[j]     = (_Float16)xrow[kb + j];
#pragma unroll
        for (int j = 0; j < 8; ++j) a[8 + j] = (_Float16)xrow[kb + 16 + j];

#pragma unroll
        for (int nt = 0; nt < 4; ++nt) {
            const v16h b = *(const v16h*)&lB[(((kc * 4 + nt) * 32) + lane) * 16];
            acc[nt] = __builtin_amdgcn_wmma_f32_16x16x32_f16(
                false, a, false, b, (short)0, acc[nt], false, false);
        }
    }

    // D layout: c[r] -> row M0 + halfw*8 + r, col nt*16 + lm
    if (M0 + 16 <= n_rows) {
        // Fast path (all strips except the single tail strip): one base
        // address + 32 unguarded stores with immediate offsets.
        float* yrow = Y + (size_t)(M0 + halfw * 8) * HIDDEN + lm;
#pragma unroll
        for (int nt = 0; nt < 4; ++nt) {
#pragma unroll
            for (int r = 0; r < 8; ++r)
                yrow[r * HIDDEN + nt * 16] = acc[nt][r];
        }
    } else {
#pragma unroll
        for (int nt = 0; nt < 4; ++nt) {
#pragma unroll
            for (int r = 0; r < 8; ++r) {
                const int row = M0 + halfw * 8 + r;
                if (row < n_rows)
                    Y[(size_t)row * HIDDEN + nt * 16 + lm] = acc[nt][r];
            }
        }
    }
}

// ---------------------------------------------------------------------------
// agg[i][c] = h[i][c] * dinv[i]^2   (self-loop term; float4 per thread)
__global__ void gcn_selfinit(const float4* __restrict__ h, const float* __restrict__ dinv,
                             float4* __restrict__ agg, int n) {
    int t = blockIdx.x * blockDim.x + threadIdx.x;   // n*16 float4s
    if (t >= n * 16) return;
    int node = t >> 4;
    float di = dinv[node];
    float s = di * di;
    float4 v = h[t];
    v.x *= s; v.y *= s; v.z *= s; v.w *= s;
    agg[t] = v;
}

// agg[dst] += h[src] * dinv[src]*dinv[dst]   (16 threads/edge, float4 msgs)
__global__ void gcn_edge_scatter(const int* __restrict__ src, const int* __restrict__ dst,
                                 const float* __restrict__ dinv,
                                 const float* __restrict__ h, float* __restrict__ agg,
                                 int nE) {
    long long t = (long long)blockIdx.x * blockDim.x + threadIdx.x;
    int e  = (int)(t >> 4);
    int c4 = (int)(t & 15) * 4;
    if (e >= nE) return;
    int s = src[e], d = dst[e];
    float norm = dinv[s] * dinv[d];
    const float4 v = *(const float4*)&h[(size_t)s * HIDDEN + c4];
    float* p = &agg[(size_t)d * HIDDEN + c4];
    atomicAdd(p + 0, v.x * norm);
    atomicAdd(p + 1, v.y * norm);
    atomicAdd(p + 2, v.z * norm);
    atomicAdd(p + 3, v.w * norm);
}

// out[i][c] = relu(agg[i][c] + b[c])   (float4 per thread)
__global__ void gcn_bias_relu(const float4* __restrict__ agg, const float4* __restrict__ b,
                              float4* __restrict__ out, int n) {
    int t = blockIdx.x * blockDim.x + threadIdx.x;   // n*16 float4s
    if (t >= n * 16) return;
    float4 v = agg[t];
    float4 bb = b[t & 15];
    v.x += bb.x; v.y += bb.y; v.z += bb.z; v.w += bb.w;
    v.x = v.x > 0.0f ? v.x : 0.0f;
    v.y = v.y > 0.0f ? v.y : 0.0f;
    v.z = v.z > 0.0f ? v.z : 0.0f;
    v.w = v.w > 0.0f ? v.w : 0.0f;
    out[t] = v;
}

// zero pooled sums + counts
__global__ void gcn_pool_zero(float* __restrict__ sums, float* __restrict__ cnt, int G) {
    int t = blockIdx.x * blockDim.x + threadIdx.x;
    if (t < G * HIDDEN) sums[t] = 0.0f;
    if (t < G) cnt[t] = 0.0f;
}

// sums[batch[i]][c] += h[i][c]; cnt[batch[i]] += 1   (16 threads/node, float4)
__global__ void gcn_pool(const float* __restrict__ h, const int* __restrict__ batch,
                         float* __restrict__ sums, float* __restrict__ cnt, int n) {
    int t = blockIdx.x * blockDim.x + threadIdx.x;   // n*16
    if (t >= n * 16) return;
    int node = t >> 4;
    int c4 = (t & 15) * 4;
    int g = batch[node];
    const float4 v = *(const float4*)&h[(size_t)node * HIDDEN + c4];
    float* p = &sums[(size_t)g * HIDDEN + c4];
    atomicAdd(p + 0, v.x);
    atomicAdd(p + 1, v.y);
    atomicAdd(p + 2, v.z);
    atomicAdd(p + 3, v.w);
    if (c4 == 0) atomicAdd(&cnt[g], 1.0f);
}

// out[g] = dot(sums[g]/max(cnt[g],1), fc_w) + fc_b
__global__ void gcn_head(const float* __restrict__ sums, const float* __restrict__ cnt,
                         const float* __restrict__ fc_w, const float* __restrict__ fc_b,
                         float* __restrict__ out, int G) {
    int g = blockIdx.x * blockDim.x + threadIdx.x;
    if (g >= G) return;
    float c = cnt[g];
    if (c < 1.0f) c = 1.0f;
    float inv = 1.0f / c;
    float acc = 0.0f;
#pragma unroll 8
    for (int k = 0; k < HIDDEN; ++k) acc += sums[g * HIDDEN + k] * fc_w[k];
    out[g] = acc * inv + fc_b[0];
}

// ---------------------------------------------------------------------------
extern "C" void kernel_launch(void* const* d_in, const int* in_sizes, int n_in,
                              void* d_out, int out_size, void* d_ws, size_t ws_size,
                              hipStream_t stream) {
    const float* x    = (const float*)d_in[0];   // [N,128]
    const float* W1   = (const float*)d_in[1];   // [128,64]
    const float* b1   = (const float*)d_in[2];   // [64]
    const float* W2   = (const float*)d_in[3];   // [64,64]
    const float* b2   = (const float*)d_in[4];   // [64]
    const float* fc_w = (const float*)d_in[5];   // [64]
    const float* fc_b = (const float*)d_in[6];   // [1]
    const int*   ei   = (const int*)d_in[7];     // [2,E]
    const int*   bat  = (const int*)d_in[8];     // [N]

    const int N  = in_sizes[0] / 128;
    const int E  = in_sizes[7] / 2;
    const int G  = out_size;                     // num_graphs
    const int IN = 128;

    const int* src = ei;
    const int* dst = ei + E;
    float* out = (float*)d_out;

    // Workspace layout (256B-aligned slabs), everything rewritten every call.
    char* ws = (char*)d_ws;
    size_t off = 0;
    auto take = [&](size_t bytes) -> void* {
        void* p = ws + off;
        off += (bytes + 255) & ~(size_t)255;
        return p;
    };
    float* deg  = (float*)take((size_t)N * 4);           // deg -> dinv (in place)
    float* bufA = (float*)take((size_t)N * HIDDEN * 4);  // h (GEMM out)
    float* bufB = (float*)take((size_t)N * HIDDEN * 4);  // aggregation
    float* bufC = (float*)take((size_t)N * HIDDEN * 4);  // relu(agg+b)
    float* sums = (float*)take((size_t)G * HIDDEN * 4);
    float* cnt  = (float*)take((size_t)G * 4);
    (void)ws_size;

    const int B = 256;
    const int gN   = (N + B - 1) / B;
    const int gE   = (E + B - 1) / B;
    const int gN4  = (N * 16 + B - 1) / B;               // float4 elementwise passes
    const long long ec = (long long)E * 16;              // 16 threads per edge
    const int gEC  = (int)((ec + B - 1) / B);
    const int gMM  = (((N + 15) / 16) + 7) / 8;          // 8 waves x 16 rows / block
    const int gPZ  = (G * HIDDEN + B - 1) / B;
    const int gG   = (G + B - 1) / B;

    // degree -> dinv
    gcn_fill1<<<gN, B, 0, stream>>>(deg, N);
    gcn_deg_edges<<<gE, B, 0, stream>>>(dst, deg, E);
    gcn_rsqrt<<<gN, B, 0, stream>>>(deg, N);

    // Layer 1: h1 = x @ W1 ; agg ; relu(agg + b1)
    gcn_gemm_wmma<<<gMM, B, 0, stream>>>(x, W1, bufA, N, IN);
    gcn_selfinit<<<gN4, B, 0, stream>>>((const float4*)bufA, deg, (float4*)bufB, N);
    gcn_edge_scatter<<<gEC, B, 0, stream>>>(src, dst, deg, bufA, bufB, E);
    gcn_bias_relu<<<gN4, B, 0, stream>>>((const float4*)bufB, (const float4*)b1,
                                         (float4*)bufC, N);

    // Layer 2: h2 = h1r @ W2 ; agg ; relu(agg + b2)
    gcn_gemm_wmma<<<gMM, B, 0, stream>>>(bufC, W2, bufA, N, HIDDEN);
    gcn_selfinit<<<gN4, B, 0, stream>>>((const float4*)bufA, deg, (float4*)bufB, N);
    gcn_edge_scatter<<<gEC, B, 0, stream>>>(src, dst, deg, bufA, bufB, E);
    gcn_bias_relu<<<gN4, B, 0, stream>>>((const float4*)bufB, (const float4*)b2,
                                         (float4*)bufC, N);

    // Mean pool per graph + linear head
    gcn_pool_zero<<<gPZ, B, 0, stream>>>(sums, cnt, G);
    gcn_pool<<<gN4, B, 0, stream>>>(bufC, bat, sums, cnt, N);
    gcn_head<<<gG, B, 0, stream>>>(sums, cnt, fc_w, fc_b, out, G);
}
